// SoftmaxRBM_70755291234985
// MI455X (gfx1250) — matrix operand order repaired and checked by reading, not compiled, not run on hardware
//
#include <hip/hip_runtime.h>

typedef __attribute__((ext_vector_type(16))) __bf16 v16bf;
typedef __attribute__((ext_vector_type(8)))  float  v8f;
typedef __attribute__((ext_vector_type(4)))  float  f4;
typedef __attribute__((ext_vector_type(4)))  unsigned int u4;

#define HID   256
#define CARD_ 16
#define POPSZ 1024
#define NVARS 4096
#define NVIS  (NVARS * CARD_)
#define ROWB  528            // LDS row stride in bytes (512 + 16 pad, 16B aligned)
#define WAVES_PER_BLK 8
#define NTILES (POPSZ / 16)  // 64 batch tiles

__device__ __forceinline__ float swap16(float v) {
  // ds_swizzle SWAPX16: xor_mask=0x10, and_mask=0x1f -> offset 0x401F
  return __int_as_float(__builtin_amdgcn_ds_swizzle(__float_as_int(v), 0x401F));
}

// prefetch one 16-row hidden tile slice for this thread into registers
__device__ __forceinline__ void load_hidden_tile(float h[16],
                                                 const float* __restrict__ hidden,
                                                 int b0, int tid) {
  const int row = tid >> 4;       // batch row within tile
  const int col = tid & 15;       // 16-float chunk along K
  const float* hp = hidden + (size_t)(b0 + row) * HID + col * 16;
  #pragma unroll
  for (int i = 0; i < 4; ++i)
    *(f4*)(h + 4 * i) = *(const f4*)(hp + 4 * i);
}

__global__ __launch_bounds__(256, 1) void rbm_softmax_kernel(
    const float* __restrict__ hidden, const float* __restrict__ W,
    const float* __restrict__ abias, float* __restrict__ out)
{
  __shared__ __align__(16) unsigned char lds[16 * ROWB];

  const int tid  = threadIdx.x;
  const int lane = tid & 31;
  const int wave = tid >> 5;
  const int half = lane >> 4;   // which 16-lane half of the wave
  const int l16  = lane & 15;

  const int v0 = (blockIdx.x * WAVES_PER_BLK + wave) * CARD_;

  // ---- A = bf16(W[v0 + l16, 0..255]) packed per WMMA 16x16x32 A layout ----
  // lane half 0: K {0..7, 16..23} of each 32-chunk; half 1: K {8..15, 24..31}
  // W is read exactly once per element -> nontemporal, keep L2 for `hidden`.
  v16bf A[8];
  {
    const float* wrow = W + (size_t)(v0 + l16) * HID;
    #pragma unroll
    for (int kc = 0; kc < 8; ++kc) {
      const int s = kc * 32 + half * 8;
      float f[16];
      *(f4*)(f + 0)  = __builtin_nontemporal_load((const f4*)(wrow + s));
      *(f4*)(f + 4)  = __builtin_nontemporal_load((const f4*)(wrow + s + 4));
      *(f4*)(f + 8)  = __builtin_nontemporal_load((const f4*)(wrow + s + 16));
      *(f4*)(f + 12) = __builtin_nontemporal_load((const f4*)(wrow + s + 20));
      v16bf t;
      #pragma unroll
      for (int i = 0; i < 16; ++i) t[i] = (__bf16)f[i];
      A[kc] = t;
    }
  }

  // bias slice for this lane's 8 output rows (m = half*8 + r)
  float bias[8];
  {
    const float* ap = abias + v0 + half * 8;
    *(f4*)(bias + 0) = *(const f4*)(ap);
    *(f4*)(bias + 4) = *(const f4*)(ap + 4);
  }

  // prefetch batch tile 0
  float h[16];
  load_hidden_tile(h, hidden, 0, tid);

  for (int bt = 0; bt < NTILES; ++bt) {
    const int b0 = bt * 16;
    __syncthreads();   // LDS free (previous iteration's readers done)

    // ---- stage prefetched tile: cvt f32->bf16, write rows into LDS ----
    {
      unsigned u[8];
      #pragma unroll
      for (int i = 0; i < 8; ++i) {
        union { __bf16 b[2]; unsigned w; } p;
        p.b[0] = (__bf16)h[2 * i];
        p.b[1] = (__bf16)h[2 * i + 1];
        u[i] = p.w;
      }
      unsigned char* dst = lds + (tid >> 4) * ROWB + (tid & 15) * 32;
      *(u4*)dst        = *(u4*)(u + 0);
      *(u4*)(dst + 16) = *(u4*)(u + 4);
    }
    __syncthreads();

    // ---- prefetch next tile while the WMMA chain runs ----
    {
      const int nb0 = (bt < NTILES - 1) ? (b0 + 16) : b0;  // last iter: benign reload
      load_hidden_tile(h, hidden, nb0, tid);
    }

    // ---- GEMM: acc[m=visible, n=batch] over K=256 via 8 WMMA ----
    // B layout (32x16 bf16): lane = n, lanes 0-15 hold K {0..7, 16..23}, lanes 16-31 K {8..15, 24..31}
    // 2-deep pipeline: issue chunk kc+1's ds_loads before wmma kc.
    v8f acc = (v8f)(0.0f);
    const unsigned char* bsrc = lds + l16 * ROWB + half * 16;
    u4 p0 = *(const u4*)(bsrc);
    u4 p1 = *(const u4*)(bsrc + 32);
    #pragma unroll
    for (int kc = 0; kc < 8; ++kc) {
      u4 n0, n1;
      if (kc < 7) {
        n0 = *(const u4*)(bsrc + (kc + 1) * 64);
        n1 = *(const u4*)(bsrc + (kc + 1) * 64 + 32);
      }
      union { u4 q[2]; v16bf v; } B;
      B.q[0] = p0;
      B.q[1] = p1;
      acc = __builtin_amdgcn_wmma_f32_16x16x32_bf16(
          false, A[kc], false, B.v, (short)0, acc, false, false);
      if (kc < 7) { p0 = n0; p1 = n1; }
    }

    // ---- softmax over the 16 visible states of each batch column ----
    // lane holds rows m = half*8 + r; partner lane (xor 16) holds the other 8
    float x[8];
    #pragma unroll
    for (int r = 0; r < 8; ++r) x[r] = acc[r] + bias[r];
    float mx = x[0];
    #pragma unroll
    for (int r = 1; r < 8; ++r) mx = fmaxf(mx, x[r]);
    mx = fmaxf(mx, swap16(mx));
    float e[8];
    float s = 0.0f;
    #pragma unroll
    for (int r = 0; r < 8; ++r) { e[r] = __expf(x[r] - mx); s += e[r]; }
    s += swap16(s);
    const float inv = __builtin_amdgcn_rcpf(s);

    // output is a 256MB write-once stream: nontemporal, don't thrash L2
    float* op = out + (size_t)(b0 + l16) * NVIS + v0 + half * 8;
    f4 o0 = { e[0] * inv, e[1] * inv, e[2] * inv, e[3] * inv };
    f4 o1 = { e[4] * inv, e[5] * inv, e[6] * inv, e[7] * inv };
    __builtin_nontemporal_store(o0, (f4*)op);
    __builtin_nontemporal_store(o1, (f4*)(op + 4));
  }
}

extern "C" void kernel_launch(void* const* d_in, const int* in_sizes, int n_in,
                              void* d_out, int out_size, void* d_ws, size_t ws_size,
                              hipStream_t stream) {
  (void)in_sizes; (void)n_in; (void)out_size; (void)d_ws; (void)ws_size;
  const float* hidden = (const float*)d_in[0];   // [1024, 256]
  const float* W      = (const float*)d_in[1];   // [65536, 256]
  const float* a      = (const float*)d_in[2];   // [65536]
  float* out          = (float*)d_out;           // [1024, 65536]

  dim3 grid(NVARS / WAVES_PER_BLK);   // 512 blocks
  dim3 block(256);                    // 8 waves, one CARD group each
  hipLaunchKernelGGL(rbm_softmax_kernel, grid, block, 0, stream,
                     hidden, W, a, out);
}
